// MultiViewReconstructionLoss_24885040513656
// MI455X (gfx1250) — matrix operand order repaired and checked by reading, not compiled, not run on hardware
//
#include <hip/hip_runtime.h>
#include <hip/hip_bf16.h>
#include <math.h>

typedef __attribute__((ext_vector_type(16))) __bf16 v16bf;
typedef __attribute__((ext_vector_type(8)))  __bf16 v8bf;
typedef __attribute__((ext_vector_type(8)))  float  v8f;
typedef __attribute__((ext_vector_type(4)))  int    v4i;

#define NB 2
#define NN 4096
#define NV 4
#define ND 256
#define NC 3
#define NH 512
#define NPAIR 12

#define TILE_M 128
#define TILE_N2 128
#define LDSTRIDE 264   // 256 bf16 + 8 pad -> 528B row stride, conflict-free b128 LDS reads

__constant__ int PAIR_I[NPAIR] = {0,0,0, 1,1,1, 2,2,2, 3,3,3};
__constant__ int PAIR_J[NPAIR] = {1,2,3, 0,2,3, 0,1,3, 0,1,2};

// ---------------------------------------------------------------------------
// CDNA5 async global->LDS staging (ASYNCcnt path), with safe fallback
// ---------------------------------------------------------------------------
#if __has_builtin(__builtin_amdgcn_global_load_async_to_lds_b128)
#define HAS_ASYNC_LDS 1
#endif

#ifdef HAS_ASYNC_LDS
typedef __attribute__((address_space(1))) v4i as1_v4i;
typedef __attribute__((address_space(3))) v4i as3_v4i;
#endif

__device__ __forceinline__ void cp16_to_lds(void* lptr, const void* gptr) {
#ifdef HAS_ASYNC_LDS
  __builtin_amdgcn_global_load_async_to_lds_b128(
      (as1_v4i*)(void*)gptr, (as3_v4i*)lptr, 0, 0);
#else
  *(uint4*)lptr = *(const uint4*)gptr;
#endif
}

__device__ __forceinline__ void cp_lds_wait() {
#ifdef HAS_ASYNC_LDS
#if __has_builtin(__builtin_amdgcn_s_wait_asynccnt)
  __builtin_amdgcn_s_wait_asynccnt(0);
#else
  asm volatile("s_wait_asynccnt 0" ::: "memory");
#endif
#endif
}

__device__ __forceinline__ float red_max16(float v) {
  #pragma unroll
  for (int d = 1; d < 16; d <<= 1) v = fmaxf(v, __shfl_xor(v, d, 32));
  return v;
}
__device__ __forceinline__ float red_sum16(float v) {
  #pragma unroll
  for (int d = 1; d < 16; d <<= 1) v += __shfl_xor(v, d, 32);
  return v;
}

// ---------------------------------------------------------------------------
// Stage A: per (b,n,v): inv_norm + bf16 convert into view-major layout
// ---------------------------------------------------------------------------
__global__ __launch_bounds__(256) void prep_desc(const float* __restrict__ desc,
                                                 __bf16* __restrict__ descB,
                                                 float* __restrict__ invn) {
  int wid  = (blockIdx.x * 256 + threadIdx.x) >> 5;   // one wave per row
  int lane = threadIdx.x & 31;
  if (wid >= NB * NN * NV) return;
  int v = wid % NV;
  int n = (wid / NV) % NN;
  int b = wid / (NV * NN);
  const float* src = desc + (((size_t)b * NN + n) * NV + v) * ND;
  __bf16* dst = descB + ((size_t)(b * NV + v) * NN + n) * ND;

  const float4* s4 = (const float4*)src;
  float4 x0 = s4[lane * 2 + 0];
  float4 x1 = s4[lane * 2 + 1];
  float ss = x0.x*x0.x + x0.y*x0.y + x0.z*x0.z + x0.w*x0.w
           + x1.x*x1.x + x1.y*x1.y + x1.z*x1.z + x1.w*x1.w;
  v8bf o;
  o[0]=(__bf16)x0.x; o[1]=(__bf16)x0.y; o[2]=(__bf16)x0.z; o[3]=(__bf16)x0.w;
  o[4]=(__bf16)x1.x; o[5]=(__bf16)x1.y; o[6]=(__bf16)x1.z; o[7]=(__bf16)x1.w;
  *(v8bf*)(dst + lane * 8) = o;

  #pragma unroll
  for (int d = 1; d < 32; d <<= 1) ss += __shfl_xor(ss, d, 32);
  if (lane == 0) {
    float nrm = sqrtf(ss);
    invn[(b * NV + v) * NN + n] = 1.f / fmaxf(nrm, 1e-12f);
  }
}

// W1 (D,H) f32 -> W1T (H,D) bf16  (K-contiguous per output column)
__global__ __launch_bounds__(256) void prep_w1(const float* __restrict__ w1,
                                               __bf16* __restrict__ w1t) {
  int t = blockIdx.x * 256 + threadIdx.x;
  if (t >= ND * NH) return;
  int k = t / NH, h = t % NH;
  w1t[(size_t)h * ND + k] = (__bf16)w1[t];
}

// per-pair tcode[p][h] = concat(T[i],T[j]) @ Wt + b1
__global__ __launch_bounds__(256) void tcode_kernel(const float* __restrict__ T,
                                                    const float* __restrict__ Wt,
                                                    const float* __restrict__ b1,
                                                    float* __restrict__ tcode) {
  int p = blockIdx.x;
  int i = PAIR_I[p], j = PAIR_J[p];
  for (int h = threadIdx.x; h < NH; h += 256) {
    float s = b1[h];
    #pragma unroll
    for (int k = 0; k < 16; ++k) s += T[i * 16 + k] * Wt[k * NH + h];
    #pragma unroll
    for (int k = 0; k < 16; ++k) s += T[j * 16 + k] * Wt[(16 + k) * NH + h];
    tcode[p * NH + h] = s;
  }
}

// ---------------------------------------------------------------------------
// Stage B: sf_recon[p][b][n][c] = relu(di@W1 + tcode_p) @ W2 + b2   (WMMA)
// ---------------------------------------------------------------------------
__global__ __launch_bounds__(256) void recon_kernel(const __bf16* __restrict__ descB,
                                                    const __bf16* __restrict__ w1t,
                                                    const float* __restrict__ tcode,
                                                    const float* __restrict__ w2,
                                                    const float* __restrict__ b2,
                                                    float* __restrict__ sf_recon) {
  __shared__ __align__(16) __bf16 ldsA[TILE_M * LDSTRIDE];
  __shared__ __align__(16) __bf16 ldsW[128 * LDSTRIDE];
  __shared__ float ldsW2[NH * NC];
  __shared__ float ldsTC[NH];

  const int p  = blockIdx.y;
  const int iv = PAIR_I[p];
  const int r0 = blockIdx.x * TILE_M;
  const int b  = r0 / NN;
  const int n0 = r0 % NN;
  const int tid = threadIdx.x, wave = tid >> 5, lane = tid & 31;
  const int lcol = lane & 15, lhalf = lane >> 4;

  const __bf16* di = descB + ((size_t)(b * NV + iv) * NN + n0) * ND;
  {
    for (int t = tid; t < TILE_M * 32; t += 256) {
      int r = t >> 5, c = t & 31;
      cp16_to_lds(ldsA + r * LDSTRIDE + c * 8, di + (size_t)r * ND + c * 8);
    }
    for (int t = tid; t < NH * NC; t += 256) ldsW2[t] = w2[t];
    for (int t = tid; t < NH; t += 256)      ldsTC[t] = tcode[p * NH + t];
  }

  float run3[8][3];
  #pragma unroll
  for (int v = 0; v < 8; ++v) { run3[v][0] = run3[v][1] = run3[v][2] = 0.f; }

  for (int hc = 0; hc < NH / 128; ++hc) {
    __syncthreads();
    for (int t = tid; t < 128 * 32; t += 256) {
      int r = t >> 5, c = t & 31;
      cp16_to_lds(ldsW + r * LDSTRIDE + c * 8,
                  w1t + ((size_t)hc * 128 + r) * ND + c * 8);
    }
    cp_lds_wait();
    __syncthreads();

    v8f acc[8];
    #pragma unroll
    for (int blk = 0; blk < 8; ++blk)
      #pragma unroll
      for (int e = 0; e < 8; ++e) acc[blk][e] = 0.f;

    #pragma unroll
    for (int kk = 0; kk < 8; ++kk) {
      const __bf16* arow = ldsA + (wave * 16 + lcol) * LDSTRIDE + kk * 32 + lhalf * 8;
      v16bf afrag;
      {
        v8bf lo = *(const v8bf*)(arow);
        v8bf hi = *(const v8bf*)(arow + 16);
        #pragma unroll
        for (int t = 0; t < 8; ++t) { afrag[t] = lo[t]; afrag[t + 8] = hi[t]; }
      }
      const int kb = kk * 32 + lhalf * 8;
      // software-pipelined B-fragment fetch (next blk in flight during WMMA)
      v8bf blo = *(const v8bf*)(ldsW + lcol * LDSTRIDE + kb);
      v8bf bhi = *(const v8bf*)(ldsW + lcol * LDSTRIDE + kb + 16);
      #pragma unroll
      for (int blk = 0; blk < 8; ++blk) {
        v16bf bfrag;
        #pragma unroll
        for (int t = 0; t < 8; ++t) { bfrag[t] = blo[t]; bfrag[t + 8] = bhi[t]; }
        if (blk < 7) {
          const __bf16* brn = ldsW + ((blk + 1) * 16 + lcol) * LDSTRIDE + kb;
          blo = *(const v8bf*)(brn);
          bhi = *(const v8bf*)(brn + 16);
        }
        acc[blk] = __builtin_amdgcn_wmma_f32_16x16x32_bf16(
            false, afrag, false, bfrag, (short)0, acc[blk], false, false);
      }
    }

    #pragma unroll
    for (int v = 0; v < 8; ++v)
      #pragma unroll
      for (int blk = 0; blk < 8; ++blk) {
        int h = hc * 128 + blk * 16 + lcol;
        float hv = fmaxf(acc[blk][v] + ldsTC[h], 0.f);
        run3[v][0] += hv * ldsW2[h * 3 + 0];
        run3[v][1] += hv * ldsW2[h * 3 + 1];
        run3[v][2] += hv * ldsW2[h * 3 + 2];
      }
  }

  #pragma unroll
  for (int v = 0; v < 8; ++v) {
    float s0 = red_sum16(run3[v][0]);
    float s1 = red_sum16(run3[v][1]);
    float s2 = red_sum16(run3[v][2]);
    if (lcol == 0) {
      int n = n0 + wave * 16 + v + 8 * lhalf;
      size_t o = ((size_t)(p * NB + b) * NN + n) * 3;
      sf_recon[o + 0] = s0 + b2[0];
      sf_recon[o + 1] = s1 + b2[1];
      sf_recon[o + 2] = s2 + b2[2];
    }
  }
}

// ---------------------------------------------------------------------------
// Stage C: flash-style S = di . dj^T with online softmax @ sf_j and online
// argmax of S * inv||dj||  (row-wise; reverse pair supplies the column argmax)
// ---------------------------------------------------------------------------
__global__ __launch_bounds__(256) void pair_kernel(const __bf16* __restrict__ descB,
                                                   const float* __restrict__ invn,
                                                   const float* __restrict__ sf,
                                                   float* __restrict__ row_soft,
                                                   int* __restrict__ row_arg) {
  __shared__ __align__(16) __bf16 ldsA[TILE_M * LDSTRIDE];
  __shared__ __align__(16) __bf16 ldsB[TILE_N2 * LDSTRIDE];
  __shared__ float ldsSF[TILE_N2 * 3];
  __shared__ float ldsInv[TILE_N2];

  const int pb = blockIdx.y;
  const int p  = pb / NB;
  const int b  = pb % NB;
  const int iv = PAIR_I[p];
  const int jv = PAIR_J[p];
  const int row0 = blockIdx.x * TILE_M;

  const __bf16* di = descB + (size_t)(b * NV + iv) * NN * ND;
  const __bf16* dj = descB + (size_t)(b * NV + jv) * NN * ND;
  const float* invj = invn + (b * NV + jv) * NN;

  const int tid = threadIdx.x, wave = tid >> 5, lane = tid & 31;
  const int lcol = lane & 15, lhalf = lane >> 4;

  for (int t = tid; t < TILE_M * 32; t += 256) {
    int r = t >> 5, c = t & 31;
    cp16_to_lds(ldsA + r * LDSTRIDE + c * 8,
                di + ((size_t)row0 + r) * ND + c * 8);
  }

  float run_max[8], run_sum[8], accc[8][3], bestv[8];
  int besti[8];
  #pragma unroll
  for (int v = 0; v < 8; ++v) {
    run_max[v] = -1e30f; run_sum[v] = 0.f;
    accc[v][0] = accc[v][1] = accc[v][2] = 0.f;
    bestv[v] = -1e30f; besti[v] = 0;
  }

  for (int mt = 0; mt < NN / TILE_N2; ++mt) {
    const int m0 = mt * TILE_N2;
    __syncthreads();
    {
      for (int t = tid; t < TILE_N2 * 32; t += 256) {
        int r = t >> 5, c = t & 31;
        cp16_to_lds(ldsB + r * LDSTRIDE + c * 8,
                    dj + ((size_t)m0 + r) * ND + c * 8);
      }
      for (int t = tid; t < TILE_N2 * 3; t += 256) {
        int m = t / 3, c = t % 3;
        ldsSF[t] = sf[(((size_t)b * NN + m0 + m) * NV + jv) * NC + c];
      }
      if (tid < TILE_N2) ldsInv[tid] = invj[m0 + tid];
    }
    cp_lds_wait();
    __syncthreads();

    // warm L2->WGP path for the next tile while this one computes
    if (mt + 1 < NN / TILE_N2)
      __builtin_prefetch(dj + ((size_t)m0 + TILE_N2) * ND + (tid & 127) * 256, 0, 1);

    v8f acc[8];
    #pragma unroll
    for (int blk = 0; blk < 8; ++blk)
      #pragma unroll
      for (int e = 0; e < 8; ++e) acc[blk][e] = 0.f;

    #pragma unroll
    for (int kk = 0; kk < 8; ++kk) {
      const __bf16* arow = ldsA + (wave * 16 + lcol) * LDSTRIDE + kk * 32 + lhalf * 8;
      v16bf afrag;
      {
        v8bf lo = *(const v8bf*)(arow);
        v8bf hi = *(const v8bf*)(arow + 16);
        #pragma unroll
        for (int t = 0; t < 8; ++t) { afrag[t] = lo[t]; afrag[t + 8] = hi[t]; }
      }
      const int kb = kk * 32 + lhalf * 8;
      v8bf blo = *(const v8bf*)(ldsB + lcol * LDSTRIDE + kb);
      v8bf bhi = *(const v8bf*)(ldsB + lcol * LDSTRIDE + kb + 16);
      #pragma unroll
      for (int blk = 0; blk < 8; ++blk) {
        v16bf bfrag;
        #pragma unroll
        for (int t = 0; t < 8; ++t) { bfrag[t] = blo[t]; bfrag[t + 8] = bhi[t]; }
        if (blk < 7) {
          const __bf16* brn = ldsB + ((blk + 1) * 16 + lcol) * LDSTRIDE + kb;
          blo = *(const v8bf*)(brn);
          bhi = *(const v8bf*)(brn + 16);
        }
        acc[blk] = __builtin_amdgcn_wmma_f32_16x16x32_bf16(
            false, afrag, false, bfrag, (short)0, acc[blk], false, false);
      }
    }

    // online epilogue: rows v + 8*lhalf of this wave's 16-row strip
    #pragma unroll
    for (int v = 0; v < 8; ++v) {
      float sv[8];
      float tmax = -1e30f;
      #pragma unroll
      for (int blk = 0; blk < 8; ++blk) { sv[blk] = acc[blk][v]; tmax = fmaxf(tmax, sv[blk]); }
      tmax = red_max16(tmax);
      float m_new = fmaxf(run_max[v], tmax);
      float scale = __expf(run_max[v] - m_new);
      float lsum = 0.f, lc0 = 0.f, lc1 = 0.f, lc2 = 0.f;
      float bv = bestv[v]; int bi = besti[v];
      #pragma unroll
      for (int blk = 0; blk < 8; ++blk) {
        int col = blk * 16 + lcol;
        float e = __expf(sv[blk] - m_new);
        lsum += e;
        lc0 += e * ldsSF[col * 3 + 0];
        lc1 += e * ldsSF[col * 3 + 1];
        lc2 += e * ldsSF[col * 3 + 2];
        float a = sv[blk] * ldsInv[col];
        int gc = m0 + col;
        if (a > bv || (a == bv && gc < bi)) { bv = a; bi = gc; }
      }
      lsum = red_sum16(lsum);
      lc0  = red_sum16(lc0);
      lc1  = red_sum16(lc1);
      lc2  = red_sum16(lc2);
      #pragma unroll
      for (int d = 1; d < 16; d <<= 1) {
        float ov = __shfl_xor(bv, d, 32);
        int   oi = __shfl_xor(bi, d, 32);
        if (ov > bv || (ov == bv && oi < bi)) { bv = ov; bi = oi; }
      }
      run_sum[v] = run_sum[v] * scale + lsum;
      accc[v][0] = accc[v][0] * scale + lc0;
      accc[v][1] = accc[v][1] * scale + lc1;
      accc[v][2] = accc[v][2] * scale + lc2;
      run_max[v] = m_new;
      bestv[v] = bv; besti[v] = bi;
    }
  }

  if (lcol == 0) {
    #pragma unroll
    for (int v = 0; v < 8; ++v) {
      int n = row0 + wave * 16 + v + 8 * lhalf;
      float is = 1.f / run_sum[v];
      size_t o = (size_t)(p * NB + b) * NN + n;
      row_soft[o * 3 + 0] = accc[v][0] * is;
      row_soft[o * 3 + 1] = accc[v][1] * is;
      row_soft[o * 3 + 2] = accc[v][2] * is;
      row_arg[o] = besti[v];
    }
  }
}

// ---------------------------------------------------------------------------
// Stage D: mutual-NN gate + per-pair MSE, then final scalar
// ---------------------------------------------------------------------------
__global__ __launch_bounds__(256) void loss_partial(const int* __restrict__ row_arg,
                                                    const float* __restrict__ sf_recon,
                                                    const float* __restrict__ row_soft,
                                                    float* __restrict__ lb,
                                                    int* __restrict__ mut) {
  __shared__ float sred[256];
  __shared__ int   ired[256];
  int pb = blockIdx.x;
  int p = pb / NB, b = pb % NB;
  int i = PAIR_I[p], j = PAIR_J[p];
  int q = j * 3 + ((i < j) ? i : i - 1);           // reverse pair index
  const int*   idxp = row_arg  + (size_t)(p * NB + b) * NN;
  const int*   idxq = row_arg  + (size_t)(q * NB + b) * NN;
  const float* rec  = sf_recon + (size_t)(p * NB + b) * NN * 3;
  const float* sft  = row_soft + (size_t)(p * NB + b) * NN * 3;
  float ss = 0.f; int found = 0;
  for (int n = threadIdx.x; n < NN; n += 256) {
    int m = idxp[n];
    if (idxq[m] == n) found = 1;
    #pragma unroll
    for (int c = 0; c < 3; ++c) { float d = rec[n * 3 + c] - sft[n * 3 + c]; ss += d * d; }
  }
  sred[threadIdx.x] = ss; ired[threadIdx.x] = found;
  __syncthreads();
  for (int s = 128; s > 0; s >>= 1) {
    if (threadIdx.x < s) {
      sred[threadIdx.x] += sred[threadIdx.x + s];
      ired[threadIdx.x] |= ired[threadIdx.x + s];
    }
    __syncthreads();
  }
  if (threadIdx.x == 0) { lb[pb] = sred[0] / (float)(NN * 3); mut[pb] = ired[0]; }
}

__global__ void loss_final(const float* __restrict__ lb,
                           const int* __restrict__ mut,
                           float* __restrict__ out) {
  int t = threadIdx.x;
  float tot = 0.f, cnt = 0.f;
  if (t < NPAIR * NB && mut[t]) { tot = lb[t]; cnt = 1.f; }
  #pragma unroll
  for (int d = 1; d < 32; d <<= 1) { tot += __shfl_xor(tot, d, 32); cnt += __shfl_xor(cnt, d, 32); }
  if (t == 0) out[0] = (cnt > 0.f) ? tot / fmaxf(cnt, 1.f) : 0.f;
}

// ---------------------------------------------------------------------------
extern "C" void kernel_launch(void* const* d_in, const int* in_sizes, int n_in,
                              void* d_out, int out_size, void* d_ws, size_t ws_size,
                              hipStream_t stream) {
  (void)in_sizes; (void)n_in; (void)out_size; (void)ws_size;
  const float* desc = (const float*)d_in[0];
  const float* sf   = (const float*)d_in[1];
  const float* T    = (const float*)d_in[2];
  const float* W1   = (const float*)d_in[3];
  const float* b1   = (const float*)d_in[4];
  const float* Wt   = (const float*)d_in[5];
  const float* W2   = (const float*)d_in[6];
  const float* b2   = (const float*)d_in[7];
  float* out = (float*)d_out;

  char* ws = (char*)d_ws;
  size_t off = 0;
  auto carve = [&](size_t bytes) -> void* {
    void* pp = ws + off;
    off = (off + bytes + 255) & ~(size_t)255;
    return pp;
  };
  __bf16* descB   = (__bf16*)carve((size_t)NB * NV * NN * ND * 2);
  float*  invn    = (float*) carve((size_t)NB * NV * NN * 4);
  __bf16* w1t     = (__bf16*)carve((size_t)NH * ND * 2);
  float*  tcode   = (float*) carve((size_t)NPAIR * NH * 4);
  float*  sfrec   = (float*) carve((size_t)NPAIR * NB * NN * 3 * 4);
  float*  rsoft   = (float*) carve((size_t)NPAIR * NB * NN * 3 * 4);
  int*    rarg    = (int*)   carve((size_t)NPAIR * NB * NN * 4);
  float*  lb      = (float*) carve((size_t)NPAIR * NB * 4);
  int*    mut     = (int*)   carve((size_t)NPAIR * NB * 4);

  prep_desc<<<(NB * NN * NV) / 8, 256, 0, stream>>>(desc, descB, invn);
  prep_w1<<<(ND * NH + 255) / 256, 256, 0, stream>>>(W1, w1t);
  tcode_kernel<<<NPAIR, 256, 0, stream>>>(T, Wt, b1, tcode);
  recon_kernel<<<dim3((NB * NN) / TILE_M, NPAIR), 256, 0, stream>>>(
      descB, w1t, tcode, W2, b2, sfrec);
  pair_kernel<<<dim3(NN / TILE_M, NPAIR * NB), 256, 0, stream>>>(
      descB, invn, sf, rsoft, rarg);
  loss_partial<<<NPAIR * NB, 256, 0, stream>>>(rarg, sfrec, rsoft, lb, mut);
  loss_final<<<1, 32, 0, stream>>>(lb, mut, out);
}
